// LineRider_55301998903628
// MI455X (gfx1250) — matrix operand order repaired and checked by reading, not compiled, not run on hardware
//
#include <hip/hip_runtime.h>
#include <hip/hip_bf16.h>
#include <math.h>

// ---------------------------------------------------------------------------
// LineRider on MI455X (gfx1250): one persistent 256-thread workgroup (8 wave32
// waves) runs the 68 sequential steps. Conv layers are implicit GEMMs on
// v_wmma_f32_16x16x32_f16 (f16 inputs, f32 accumulation).
//
// Per conv layer (per round of N-tiles):
//   - all 256 threads build a zero-padded f16 im2col panel in LDS
//     (column-major per output pixel, +8 f16 column pad for bank stagger)
//   - each wave consumes one (N-tile, M-tile) pair:
//       B = 2x ds_load_b128 (f16 panel), A = 2x global_load_b128 (f16 weights
//       pre-converted/padded into d_ws by a one-time prep pass), then WMMA.
//   The f16-weight vs f32-weight (no-workspace fallback) choice is a template
//   parameter selected once per layer on a kernel-arg-uniform flag, so the
//   hot loop carries no per-tile checks.
// All activations live in LDS (~290 KB of the 320 KB WGP LDS).
//
// Input order (setup_inputs dict order; params dict flattened in decl order):
//  0 img[3*4096*4096] f32   1 box_size (int)   2 x_0   3 y_0   4 angle_0
//  5 c1_w  6 c1_b  7 c2_w  8 c2_b  9 c3_w 10 c3_b 11 c4_w 12 c4_b
// 13 c5_w 14 c5_b 15 c6_w 16 c6_b 17 c7_w 18 c7_b
// 19 fc_w 20 fc_b 21 out_w 22 out_b 23 hid_w 24 hid_b
// Output: concat(points[(S+1)*2], bl_end[S+1], bl_len[S+1]) f32, S = W/box+4.
// ---------------------------------------------------------------------------

typedef __attribute__((ext_vector_type(16))) _Float16 v16h;
typedef __attribute__((ext_vector_type(8)))  _Float16 v8h;
typedef __attribute__((ext_vector_type(2)))  _Float16 v2h;
typedef __attribute__((ext_vector_type(8)))  float    v8f;
typedef __attribute__((ext_vector_type(4)))  float    v4f;

#define NWAVES 8   // blockDim.x == 256, wave32

// LDS layout (float offsets)
#define ST_OFF     0                       // state scratch (64 floats)
#define PATCH_OFF  64                      // 3*32*32
#define FEAT_OFF   (PATCH_OFF + 3072)      // 128
#define BUF0_OFF   (FEAT_OFF + 128)        // max 32*30*30 = 28800 f32
#define BUF1_OFF   (BUF0_OFF + 28800)      // max 32*32*32 = 32768 f32
#define PANEL_OFF  (BUF1_OFF + 32768)      // max 18944 f16 = 9472 f32
#define SMEM_FLOATS (PANEL_OFF + 9472)     // 74304 floats = 297216 B (<320KB)

// state scratch indices
#define S_X 0
#define S_Y 1
#define S_ANG 2
#define S_SINA 3
#define S_COSA 4
#define S_HID 5        // 8
#define S_TH 13        // 6 (theta row-major 2x3)
#define S_CNN 19       // 8
#define S_CMB 27       // 16
#define S_OUT 43       // 4

// ---------------------------------------------------------------------------
// Weight prep: f32 [COUT][K] -> f16 [COUT][KP] (K zero-padded to mult. of 32).
// ---------------------------------------------------------------------------
__global__ void prep_weights_f16(const float* __restrict__ src,
                                 _Float16* __restrict__ dst,
                                 int COUT, int K, int KP)
{
    const int n = COUT * KP;
    for (int i = blockIdx.x * blockDim.x + threadIdx.x; i < n;
         i += gridDim.x * blockDim.x) {
        const int m = i / KP, kp = i % KP;
        dst[i] = (kp < K) ? (_Float16)src[m * K + kp] : (_Float16)0.0f;
    }
}

// ---------------------------------------------------------------------------
// 3x3 conv as implicit GEMM on WMMA.  O[co,p] = sum_k W[co,k] * im2col[k,p],
// k = ci*9 + ky*3 + kx, p = oy*WOUT+ox.  Bias + ReLU fused into the store.
// ISA wave32 fragment layouts (g = lane>>4, l = lane&15):
//   A(16x32,f16): M = l; half h<8 -> K=8g+h, h>=8 -> K=16+8g+(h-8)
//   B(32x16,f16): N = l; same K order as A
//   C(16x16,f32): acc[r] -> (M = mBase+r+8g, N = nBase+l)
// F16W: compile-time weight-source select (f16 prepped in d_ws vs f32+cvt).
// ---------------------------------------------------------------------------
template <int CIN, int COUT, int HIN, int WIN, int PAD, bool F16W>
__device__ void conv3x3_wmma(const float* __restrict__ Wt,
                             const _Float16* __restrict__ Wf,
                             const float* __restrict__ Bs,
                             const float* __restrict__ actIn,
                             float* __restrict__ actOut,
                             _Float16* __restrict__ panel,
                             int tid, int nT, int lane, int wave)
{
    constexpr int HOUT = HIN + 2 * PAD - 2;
    constexpr int WOUT = WIN + 2 * PAD - 2;
    constexpr int NPIX = HOUT * WOUT;
    constexpr int K    = CIN * 9;
    constexpr int KC   = (K + 31) / 32;      // K chunks of 32
    constexpr int KP   = KC * 32;            // zero-padded K
    constexpr int COL  = KP + 8;             // f16 per panel column (bank pad)
    constexpr int MT   = COUT / 16;          // <= NWAVES for all layers
    constexpr int NT   = (NPIX + 15) / 16;
    constexpr int WCHUNK = (NWAVES / MT) < 1 ? 1 : (NWAVES / MT);
    constexpr int ROUNDS = (NT + WCHUNK - 1) / WCHUNK;

    const int g = lane >> 4;
    const int l = lane & 15;
    const int cIdx  = wave / MT;             // which N-tile of this round
    const int mt    = wave % MT;
    const int mBase = mt * 16;

    for (int rnd = 0; rnd < ROUNDS; ++rnd) {
        const int ntBase = rnd * WCHUNK;

        // ---- build f16 im2col panels (all threads; paired f16 stores) ----
        constexpr int HALF = KP / 2;
        for (int e = tid; e < WCHUNK * 16 * HALF; e += nT) {
            const int c  = e / (16 * HALF);
            const int r0 = e % (16 * HALF);
            const int j  = r0 / HALF;
            const int kp = (r0 % HALF) * 2;
            const int n  = (ntBase + c) * 16 + j;
            float v0 = 0.0f, v1 = 0.0f;
            if (n < NPIX) {
                const int oy = n / WOUT, ox = n % WOUT;
#pragma unroll
                for (int u = 0; u < 2; ++u) {
                    const int k = kp + u;
                    if (k < K) {
                        const int ci = k / 9, rr = k % 9;
                        const int iy = oy + rr / 3 - PAD;
                        const int ix = ox + rr % 3 - PAD;
                        if (iy >= 0 && iy < HIN && ix >= 0 && ix < WIN) {
                            const float t = actIn[(ci * HIN + iy) * WIN + ix];
                            if (u == 0) v0 = t; else v1 = t;
                        }
                    }
                }
            }
            v2h pk; pk[0] = (_Float16)v0; pk[1] = (_Float16)v1;
            *reinterpret_cast<v2h*>(&panel[(c * 16 + j) * COL + kp]) = pk;
        }
        __syncthreads();

        // ---- consume: each wave one (N-tile, M-tile) pair ----
        const int ntv = ntBase + cIdx;
        if (cIdx < WCHUNK && ntv < NT) {
            const int nBase = ntv * 16;
            const int m = mBase + l;
            const int p = nBase + l;
            const _Float16* colp = panel + (cIdx * 16 + l) * COL;

            v8f acc = {};
            for (int k0 = 0; k0 < KP; k0 += 32) {
                v16h a;
                if constexpr (F16W) {
                    // fast path: f16 pre-padded weights, 2x b128 per WMMA
                    const _Float16* wrow = Wf + m * KP;
                    const v8h a0 = *reinterpret_cast<const v8h*>(wrow + k0 + 8 * g);
                    const v8h a1 = *reinterpret_cast<const v8h*>(wrow + k0 + 8 * g + 16);
#pragma unroll
                    for (int h = 0; h < 8; ++h) { a[h] = a0[h]; a[h + 8] = a1[h]; }
                } else if constexpr (K % 32 == 0) {
                    // fallback: f32 weights + convert (no workspace available)
                    const v4f* wp0 = reinterpret_cast<const v4f*>(Wt + m * K + k0 + 8 * g);
                    const v4f* wp1 = reinterpret_cast<const v4f*>(Wt + m * K + k0 + 8 * g + 16);
                    const v4f w00 = wp0[0], w01 = wp0[1];
                    const v4f w10 = wp1[0], w11 = wp1[1];
#pragma unroll
                    for (int h = 0; h < 4; ++h) {
                        a[h]      = (_Float16)w00[h];
                        a[h + 4]  = (_Float16)w01[h];
                        a[h + 8]  = (_Float16)w10[h];
                        a[h + 12] = (_Float16)w11[h];
                    }
                } else {
#pragma unroll
                    for (int h = 0; h < 8; ++h) {
                        const int k = k0 + 8 * g + h;
                        float w0 = 0.0f, w1 = 0.0f;
                        if (k < K)      w0 = Wt[m * K + k];
                        if (k + 16 < K) w1 = Wt[m * K + k + 16];
                        a[h]     = (_Float16)w0;
                        a[h + 8] = (_Float16)w1;
                    }
                }

                // B: K = 8g+h and 16+8g+h from the panel (2x ds_load_b128)
                const v8h b0 = *reinterpret_cast<const v8h*>(colp + k0 + 8 * g);
                const v8h b1 = *reinterpret_cast<const v8h*>(colp + k0 + 8 * g + 16);
                v16h b;
#pragma unroll
                for (int h = 0; h < 8; ++h) { b[h] = b0[h]; b[h + 8] = b1[h]; }

                acc = __builtin_amdgcn_wmma_f32_16x16x32_f16(
                    false, a, false, b, (short)0, acc, false, false);
            }

            if (p < NPIX) {
                const int oy = p / WOUT, ox = p % WOUT;
#pragma unroll
                for (int r = 0; r < 8; ++r) {          // C store + bias + ReLU
                    const int co = mBase + r + 8 * g;
                    actOut[(co * HOUT + oy) * WOUT + ox] =
                        fmaxf(acc[r] + Bs[co], 0.0f);
                }
            }
        }
        __syncthreads();
    }
}

// 2x2 max pool, stride 2, square maps
template <int C, int H>
__device__ void pool2x2(const float* __restrict__ in, float* __restrict__ out,
                        int tid, int nT)
{
    constexpr int HO = H / 2;
    for (int i = tid; i < C * HO * HO; i += nT) {
        const int c = i / (HO * HO);
        const int r = i % (HO * HO);
        const int y = r / HO, x = r % HO;
        const float* b = in + (c * H + 2 * y) * H + 2 * x;
        out[i] = fmaxf(fmaxf(b[0], b[1]), fmaxf(b[H], b[H + 1]));
    }
}

__global__ __launch_bounds__(256) void linerider_kernel(
    const float* __restrict__ img, const int* __restrict__ boxp,
    const float* __restrict__ x0p, const float* __restrict__ y0p,
    const float* __restrict__ a0p,
    const float* c1w, const float* c1b, const float* c2w, const float* c2b,
    const float* c3w, const float* c3b, const float* c4w, const float* c4b,
    const float* c5w, const float* c5b, const float* c6w, const float* c6b,
    const float* c7w, const float* c7b, const float* fcw, const float* fcb,
    const float* outw, const float* outb, const float* hidw, const float* hidb,
    const _Float16* w1f, const _Float16* w2f, const _Float16* w3f,
    const _Float16* w4f, const _Float16* w5f, const _Float16* w6f,
    const _Float16* w7f,
    float* __restrict__ dout, int imgH, int imgW)
{
    extern __shared__ float smem[];
    float*     ST    = smem + ST_OFF;
    float*     PATCH = smem + PATCH_OFF;
    float*     FEAT  = smem + FEAT_OFF;
    float*     BUF0  = smem + BUF0_OFF;
    float*     BUF1  = smem + BUF1_OFF;
    _Float16*  PANEL = reinterpret_cast<_Float16*>(smem + PANEL_OFF);

    const int tid  = threadIdx.x;
    const int nT   = blockDim.x;
    const int lane = tid & 31;
    const int wave = tid >> 5;

    const int   box  = *boxp;
    const float boxf = (float)box;
    const int   S    = imgW / box + 4;
    const int   nOut = S + 1;
    const float Wf = (float)imgW, Hf = (float)imgH;
    const float scale_x = (2.0f * boxf) / Wf;   // box_width / img_w
    const float scale_y = boxf / Hf;            // box_height / img_h
    const float wb2     = (2.0f * boxf) / Wf;   // w_box / 2
    const float hb4     = boxf / (2.0f * Hf);   // h_box / 4

    const bool useF16 = (w1f != nullptr);       // kernel-arg uniform

    // ---- init state & constant outputs ----
    if (tid == 0) {
        const float a0 = *a0p;
        ST[S_X] = *x0p; ST[S_Y] = *y0p; ST[S_ANG] = a0;
        ST[S_SINA] = sinf(a0); ST[S_COSA] = cosf(a0);
        dout[0] = *x0p; dout[1] = *y0p;
        dout[2 * nOut] = 0.0f;           // bl_end[0]
        dout[3 * nOut] = 0.0f;           // bl_len[0]
    }
    if (tid < 8) ST[S_HID + tid] = 0.0f;
    __syncthreads();

    for (int s = 0; s < S; ++s) {
        // ---- theta (thread 0) ----
        if (tid == 0) {
            const float x = ST[S_X], y = ST[S_Y], alpha = ST[S_ANG];
            const float xs = -1.0f + x / Wf * 2.0f + wb2 * cosf(alpha);
            const float ys = -1.0f + y / Hf * 2.0f - wb2 * sinf(alpha) - hb4;
            const float ca = cosf(-alpha), sa = sinf(-alpha);
            ST[S_TH + 0] = ca * scale_x; ST[S_TH + 1] = -sa * scale_y; ST[S_TH + 2] = xs;
            ST[S_TH + 3] = sa * scale_x; ST[S_TH + 4] =  ca * scale_y; ST[S_TH + 5] = ys;
        }
        __syncthreads();

        // ---- grid_sample(nearest, zeros, align_corners=False) -> PATCH ----
        {
            const float t00 = ST[S_TH + 0], t01 = ST[S_TH + 1], t02 = ST[S_TH + 2];
            const float t10 = ST[S_TH + 3], t11 = ST[S_TH + 4], t12 = ST[S_TH + 5];
            for (int i = tid; i < 3 * 32 * 32; i += nT) {
                const int c = i >> 10, h = (i >> 5) & 31, w = i & 31;
                const float X = (w + 0.5f) * 0.0625f - 1.0f;
                const float Y = (h + 0.5f) * 0.0625f - 1.0f;
                const float gx = X * t00 + Y * t01 + t02;
                const float gy = X * t10 + Y * t11 + t12;
                const int ixr = (int)rintf(((gx + 1.0f) * Wf - 1.0f) * 0.5f);
                const int iyr = (int)rintf(((gy + 1.0f) * Hf - 1.0f) * 0.5f);
                const bool valid = (ixr >= 0) & (ixr < imgW) & (iyr >= 0) & (iyr < imgH);
                const int ixc = min(max(ixr, 0), imgW - 1);
                const int iyc = min(max(iyr, 0), imgH - 1);
                const float v = img[((size_t)c * imgH + iyc) * imgW + ixc];
                PATCH[i] = valid ? v : 0.0f;
            }
        }
        __syncthreads();

        // ---- CNN (activations in LDS, ping-pong BUF0/BUF1) ----
        // Per-layer uniform branch selects the compile-time weight path; legal
        // around __syncthreads since every thread takes the same side.
#define CONV(CI, CO, HI, WI, PD, wt, wfp, bs, ain, aout)                        \
        do {                                                                    \
            if (useF16)                                                         \
                conv3x3_wmma<CI, CO, HI, WI, PD, true >(wt, wfp, bs, ain, aout, \
                    PANEL, tid, nT, lane, wave);                                \
            else                                                                \
                conv3x3_wmma<CI, CO, HI, WI, PD, false>(wt, wfp, bs, ain, aout, \
                    PANEL, tid, nT, lane, wave);                                \
        } while (0)

        CONV( 3, 32, 32, 32, 0, c1w, w1f, c1b, PATCH, BUF0);   // 32x30x30
        CONV(32, 32, 30, 30, 2, c2w, w2f, c2b, BUF0, BUF1);    // 32x32x32
        pool2x2<32, 32>(BUF1, BUF0, tid, nT);                  // 32x16x16
        __syncthreads();
        CONV(32, 64, 16, 16, 0, c3w, w3f, c3b, BUF0, BUF1);    // 64x14x14
        CONV(64, 64, 14, 14, 2, c4w, w4f, c4b, BUF1, BUF0);    // 64x16x16
        pool2x2<64, 16>(BUF0, BUF1, tid, nT);                  // 64x8x8
        __syncthreads();
        CONV( 64, 128,  8,  8, 0, c5w, w5f, c5b, BUF1, BUF0);  // 128x6x6
        CONV(128, 128,  6,  6, 0, c6w, w6f, c6b, BUF0, BUF1);  // 128x4x4
        CONV(128, 128,  4,  4, 0, c7w, w7f, c7b, BUF1, BUF0);  // 128x2x2
        pool2x2<128, 2>(BUF0, FEAT, tid, nT);                  // 128
        __syncthreads();
#undef CONV

        // ---- FC tail ----
        if (tid < 8) {                               // cnn_out = feat @ fc_w.T + fc_b
            float acc = fcb[tid];
            for (int k = 0; k < 128; ++k) acc += FEAT[k] * fcw[tid * 128 + k];
            ST[S_CNN + tid] = acc;
        }
        __syncthreads();
        if (tid < 16)                                // combined = [cnn_out, hidden]
            ST[S_CMB + tid] = (tid < 8) ? ST[S_CNN + tid] : ST[S_HID + tid - 8];
        __syncthreads();
        if (tid < 8) {                               // hidden_new (overwrites hidden)
            float acc = hidb[tid];
            for (int k = 0; k < 16; ++k) acc += ST[S_CMB + k] * hidw[tid * 16 + k];
            ST[S_HID + tid] = acc;
        }
        if (tid < 4) {                               // out = combined @ out_w.T + out_b
            float acc = outb[tid];
            for (int k = 0; k < 16; ++k) acc += ST[S_CMB + k] * outw[tid * 16 + k];
            ST[S_OUT + tid] = acc;
        }
        __syncthreads();

        // ---- recurrence update + output writes (thread 0) ----
        if (tid == 0) {
            const float sina_o = tanhf(ST[S_OUT + 0]);
            const float cosa_o = tanhf(ST[S_OUT + 1]);
            const float bl_end = 1.0f / (1.0f + expf(-ST[S_OUT + 2]));
            const float bl_len = 1.0f / (1.0f + expf(-ST[S_OUT + 3]));
            const float nrm = sqrtf(sina_o * sina_o + cosa_o * cosa_o);
            const float sn = sina_o / nrm, cn = cosa_o / nrm;
            const float x = ST[S_X], y = ST[S_Y];
            const float sina = ST[S_SINA], cosa = ST[S_COSA];
            const float xn = x + boxf * (cosa * cn - sina * sn);
            const float yn = y - boxf * (sina * cn + cosa * sn);
            ST[S_X] = xn; ST[S_Y] = yn;
            ST[S_ANG] = ST[S_ANG] + atanf(sn / cn);
            ST[S_SINA] = sn; ST[S_COSA] = cn;
            dout[2 * (s + 1)]        = xn;
            dout[2 * (s + 1) + 1]    = yn;
            dout[2 * nOut + (s + 1)] = bl_end;
            dout[3 * nOut + (s + 1)] = bl_len;
        }
        __syncthreads();
    }
}

extern "C" void kernel_launch(void* const* d_in, const int* in_sizes, int n_in,
                              void* d_out, int out_size, void* d_ws, size_t ws_size,
                              hipStream_t stream)
{
    (void)n_in; (void)out_size;

    // img is [1,3,H,W] with H==W: recover the side from the flat size.
    const long long imgElems = (long long)in_sizes[0] / 3;
    int side = (int)(sqrt((double)imgElems) + 0.5);

    const float* img  = (const float*)d_in[0];
    const int*   boxp = (const int*)  d_in[1];
    const float* x0p  = (const float*)d_in[2];
    const float* y0p  = (const float*)d_in[3];
    const float* a0p  = (const float*)d_in[4];

    // f16 weight prep: layer (COUT, K, KP) and f16-element offsets in d_ws.
    static const int LC[7][3] = {
        {32,   27,   32}, {32,  288,  288}, {64, 288, 288}, {64, 576, 576},
        {128, 576,  576}, {128, 1152, 1152}, {128, 1152, 1152}};
    size_t offs[7], total = 0;
    for (int i = 0; i < 7; ++i) { offs[i] = total; total += (size_t)LC[i][0] * LC[i][2]; }
    const bool useF16W = (ws_size >= total * sizeof(_Float16));

    _Float16* wf[7] = {nullptr, nullptr, nullptr, nullptr, nullptr, nullptr, nullptr};
    if (useF16W) {
        _Float16* base = (_Float16*)d_ws;
        for (int i = 0; i < 7; ++i) {
            wf[i] = base + offs[i];
            const int n = LC[i][0] * LC[i][2];
            prep_weights_f16<<<(n + 255) / 256, 256, 0, stream>>>(
                (const float*)d_in[5 + 2 * i], wf[i], LC[i][0], LC[i][1], LC[i][2]);
        }
    }

    const size_t smemBytes = (size_t)SMEM_FLOATS * sizeof(float); // ~290 KB (WGP LDS 320 KB)

    linerider_kernel<<<1, 256, smemBytes, stream>>>(
        img, boxp, x0p, y0p, a0p,
        (const float*)d_in[5],  (const float*)d_in[6],
        (const float*)d_in[7],  (const float*)d_in[8],
        (const float*)d_in[9],  (const float*)d_in[10],
        (const float*)d_in[11], (const float*)d_in[12],
        (const float*)d_in[13], (const float*)d_in[14],
        (const float*)d_in[15], (const float*)d_in[16],
        (const float*)d_in[17], (const float*)d_in[18],
        (const float*)d_in[19], (const float*)d_in[20],
        (const float*)d_in[21], (const float*)d_in[22],
        (const float*)d_in[23], (const float*)d_in[24],
        wf[0], wf[1], wf[2], wf[3], wf[4], wf[5], wf[6],
        (float*)d_out, side, side);
}
